// SelfAttention_44272522887765
// MI455X (gfx1250) — matrix operand order repaired and checked by reading, not compiled, not run on hardware
//
#include <hip/hip_runtime.h>
#include <hip/hip_bf16.h>

// ---------------- problem constants ----------------
#define Bc   4
#define Lc   1024
#define DIMc 2048
#define NHc  16
#define HDc  128
#define EPSc 1e-6f
#define ROWS (Bc * Lc)                   // 4096 token rows
#define SCALEc 0.08838834764831845f     // 1/sqrt(128)

typedef __attribute__((ext_vector_type(16))) _Float16 v16h;
typedef __attribute__((ext_vector_type(8)))  _Float16 h8;
typedef __attribute__((ext_vector_type(8)))  float    v8f;
typedef int async_v4i __attribute__((vector_size(16)));   // matches builtin param

union F16x16 { v16h v; h8 h[2]; };

static __device__ __forceinline__ v8f wmma_f16(v16h a, v16h b, v8f c) {
  // D = A(16x32 f16) * B(32x16 f16) + C(16x16 f32)
  return __builtin_amdgcn_wmma_f32_16x16x32_f16(false, a, false, b, (short)0, c, false, false);
}

// ------- CDNA5 async global->LDS copy (ASYNCcnt-tracked) -------
// Probe-confirmed: this toolchain's builtin takes (global int4*, local int4*,
// imm offset, imm cpol). LDS pointer is formed from the low 32 bits of the
// flat address (== LDS byte offset per the aperture rules) via integer detour.
#if __has_builtin(__builtin_amdgcn_global_load_async_to_lds_b128)
#define ASYNC_B128(ldst, gsrc)                                                   \
  __builtin_amdgcn_global_load_async_to_lds_b128(                                \
      (__attribute__((address_space(1))) async_v4i*)(unsigned long long)(gsrc),  \
      (__attribute__((address_space(3))) async_v4i*)(unsigned long long)(ldst),  \
      0, 0)
#else
#define ASYNC_B128(ldst, gsrc)                                                   \
  asm volatile("global_load_async_to_lds_b128 %0, %1, off"                       \
               :: "v"((unsigned)(unsigned long long)(ldst)),                     \
                  "v"((unsigned long long)(gsrc))                                \
               : "memory")
#endif

#if __has_builtin(__builtin_amdgcn_s_wait_asynccnt)
#define WAIT_ASYNC(n) __builtin_amdgcn_s_wait_asynccnt(n)
#else
#define WAIT_ASYNC(n) asm volatile("s_wait_asynccnt %0" :: "i"(n) : "memory")
#endif

// ---------------- 1) W[n][k] f32 -> Wt[k][n] f16 ----------------
__global__ void __launch_bounds__(256)
wtrans_kernel(const float* __restrict__ W, _Float16* __restrict__ Wt) {
  size_t i = (size_t)blockIdx.x * 256 + threadIdx.x;   // coalesced read over k
  int n = (int)(i >> 11);
  int k = (int)(i & 2047);
  Wt[(size_t)k * DIMc + n] = (_Float16)W[i];
}

// ---------------- 2) x f32 -> f16 ----------------
__global__ void __launch_bounds__(256)
convert_x_kernel(const float* __restrict__ X, _Float16* __restrict__ Xh) {
  size_t i = (size_t)blockIdx.x * 256 + threadIdx.x;
  Xh[i] = (_Float16)X[i];
}

// ---------------- 3) Y f32 = A_f16 @ Wt + bias  (LDS-staged, async) --------
// 4 waves per block; block tile M=32 x N=128; K chunks of 32.
// Wt chunk (32 x 128 f16) double-buffered in LDS, filled with async b128 loads.
__global__ void __launch_bounds__(128)
wmma_gemm_kernel(const _Float16* __restrict__ A,   // [ROWS][DIM] (K contiguous)
                 const _Float16* __restrict__ Bt,  // [DIM(k)][DIM(n)] (N contiguous)
                 const float* __restrict__ bias,   // [DIM]
                 float* __restrict__ Y) {          // [ROWS][DIM]
  __shared__ __align__(32) _Float16 Bs[2][32][128];   // 16 KB double buffer

  const int NT = DIMc / 128;                          // 16 n-blocks
  const int m0 = (blockIdx.x / NT) * 32;
  const int n0 = (blockIdx.x % NT) * 128;
  const int tid  = threadIdx.x;
  const int lane = tid & 31;
  const int wave = tid >> 5;                          // 0..3
  const int mw = wave >> 1;                           // m-half of block tile
  const int nw = wave & 1;                            // n-half (64 wide)
  const int row = lane & 15;
  const int hi  = lane >> 4;

  v8f acc0 = {}, acc1 = {}, acc2 = {}, acc3 = {};
  const _Float16* arow = A + (size_t)(m0 + mw * 16 + row) * DIMc;

  // stage one 32x128 chunk of Bt into Bs[buf] (512 x 16B segments, 4/thread)
#define GEMM_STAGE(buf, k0)                                                    \
  _Pragma("unroll")                                                            \
  for (int t = 0; t < 4; ++t) {                                                \
    int s = tid + t * 128;                                                     \
    int r = s >> 4, c = s & 15;                                                \
    ASYNC_B128(&Bs[buf][r][c * 8],                                             \
               Bt + (size_t)((k0) + r) * DIMc + n0 + c * 8);                   \
  }

  const int NC = DIMc / 32;                           // 64 K chunks
  GEMM_STAGE(0, 0)
  for (int kc = 0; kc < NC; ++kc) {
    const int buf = kc & 1;
    const int k0 = kc * 32;
    if (kc + 1 < NC) { GEMM_STAGE(buf ^ 1, k0 + 32) WAIT_ASYNC(4); }
    else             { WAIT_ASYNC(0); }
    __syncthreads();

    // A fragment from global: lane(row) holds K {0..7,16..23} / {8..15,24..31}
    F16x16 a;
    a.h[0] = *(const h8*)(arow + k0 + hi * 8);
    a.h[1] = *(const h8*)(arow + k0 + 16 + hi * 8);
    // B fragments from LDS: lane = N column, K {0..7,16..23} / {8..15,24..31}
    const _Float16* bk = &Bs[buf][0][nw * 64 + row];
    v16h b0, b1, b2, b3;
#pragma unroll
    for (int s = 0; s < 16; ++s) {
      const int kk = (s < 8) ? hi * 8 + s : 16 + hi * 8 + (s - 8);
      b0[s] = bk[kk * 128];
      b1[s] = bk[kk * 128 + 16];
      b2[s] = bk[kk * 128 + 32];
      b3[s] = bk[kk * 128 + 48];
    }
    acc0 = wmma_f16(a.v, b0, acc0);
    acc1 = wmma_f16(a.v, b1, acc1);
    acc2 = wmma_f16(a.v, b2, acc2);
    acc3 = wmma_f16(a.v, b3, acc3);
    __syncthreads();
  }
#undef GEMM_STAGE

  // C layout: vgpr r -> M = hi*8 + r, column N = (lane&15) per 16-wide tile
  const int col = lane & 15;
  const int nb = n0 + nw * 64;
  float bv0 = bias[nb + col];
  float bv1 = bias[nb + 16 + col];
  float bv2 = bias[nb + 32 + col];
  float bv3 = bias[nb + 48 + col];
#pragma unroll
  for (int r = 0; r < 8; ++r) {
    int m = m0 + mw * 16 + hi * 8 + r;
    float* yr = Y + (size_t)m * DIMc + nb;
    yr[col]      = acc0[r] + bv0;
    yr[16 + col] = acc1[r] + bv1;
    yr[32 + col] = acc2[r] + bv2;
    yr[48 + col] = acc3[r] + bv3;
  }
}

// ---------------- 4) RMS-norm + 2D RoPE, write f16 Q or K^T ----------------
__global__ void __launch_bounds__(256)
rmsnorm_rope_kernel(const float* __restrict__ Y, const float* __restrict__ g,
                    const float* __restrict__ fcos, const float* __restrict__ fsin,
                    _Float16* __restrict__ out, int kt_transposed) {
  __shared__ float red[256];
  const int rowi = blockIdx.x;           // 0..4095
  const int b = rowi >> 10;
  const int l = rowi & 1023;
  const int hh = l >> 5;                 // grid h position (H=32)
  const int ww = l & 31;                 // grid w position (W=32)
  const float* y = Y + (size_t)rowi * DIMc;
  const int t = threadIdx.x;

  float v[8];
  float ss = 0.f;
#pragma unroll
  for (int i = 0; i < 8; ++i) { v[i] = y[t * 8 + i]; ss += v[i] * v[i]; }
  red[t] = ss;
  __syncthreads();
  for (int s = 128; s > 0; s >>= 1) {
    if (t < s) red[t] += red[t + s];
    __syncthreads();
  }
  const float rstd = rsqrtf(red[0] * (1.0f / (float)DIMc) + EPSc);

#pragma unroll
  for (int pi = 0; pi < 4; ++pi) {
    const int e = t * 8 + pi * 2;        // even element index in [0,2048)
    const int h = e >> 7;                // head
    const int d = e & 127;               // channel within head
    const int p = d >> 1;                // rope pair channel 0..63
    const int pos = (p < 32) ? hh : ww;  // first 32 chans: h-pos, rest: w-pos
    const float c  = fcos[pos * 64 + p];
    const float s_ = fsin[pos * 64 + p];
    const float xr = v[pi * 2]     * rstd * g[e];
    const float xi = v[pi * 2 + 1] * rstd * g[e + 1];
    const float orr = xr * c - xi * s_;
    const float oii = xr * s_ + xi * c;
    if (!kt_transposed) {                // Q: [B,NH,L,HD]
      _Float16* o = out + (((size_t)(b * NHc + h) * Lc + l) * HDc + d);
      o[0] = (_Float16)orr;
      o[1] = (_Float16)oii;
    } else {                             // K^T: [B,NH,HD,L]
      out[((size_t)(b * NHc + h) * HDc + d)     * Lc + l] = (_Float16)orr;
      out[((size_t)(b * NHc + h) * HDc + d + 1) * Lc + l] = (_Float16)oii;
    }
  }
}

// ---------------- 5) V reorder: Y[B*L][DIM] f32 -> V[B,NH,L,HD] f16 --------
__global__ void __launch_bounds__(256)
reorder_v_kernel(const float* __restrict__ Y, _Float16* __restrict__ V) {
  size_t i = (size_t)blockIdx.x * 256 + threadIdx.x;
  const int rowi = (int)(i >> 11);
  const int e = (int)(i & 2047);
  const int b = rowi >> 10, l = rowi & 1023;
  const int h = e >> 7, d = e & 127;
  V[((size_t)(b * NHc + h) * Lc + l) * HDc + d] = (_Float16)Y[i];
}

// ---------------- 6) flash attention (4 waves share K/V tiles in LDS) ------
// Block: 128 threads = 4 waves, 64 queries of one (b,h); 32-key steps.
// K^T tile (128 hd x 32 keys) and V tile (32 keys x 128 hd) double-buffered
// in LDS via async b128 loads; each wave computes its own 16-query tile.
__global__ void __launch_bounds__(128)
flash_attn_kernel(const _Float16* __restrict__ Q,   // [B*NH, L, HD]
                  const _Float16* __restrict__ Kt,  // [B*NH, HD, L]
                  const _Float16* __restrict__ V,   // [B*NH, L, HD]
                  _Float16* __restrict__ O) {       // [B*L, DIM] token-major
  __shared__ __align__(32) _Float16 Ks[2][128][32];  // 16 KB
  __shared__ __align__(32) _Float16 Vs[2][32][128];  // 16 KB
  __shared__ __align__(32) _Float16 Ps[4][16][32];   // per-wave prob tiles

  const int QT = Lc / 64;                            // 16 q-blocks per (b,h)
  const int bh  = blockIdx.x / QT;
  const int wq0 = (blockIdx.x % QT) * 64;
  const int tid  = threadIdx.x;
  const int lane = tid & 31;
  const int wave = tid >> 5;
  const int q0 = wq0 + wave * 16;
  const int row = lane & 15;
  const int hi  = lane >> 4;
  const int col = row;

  const _Float16* Qb  = Q  + (size_t)bh * Lc * HDc;
  const _Float16* Ktb = Kt + (size_t)bh * HDc * Lc;
  const _Float16* Vb  = V  + (size_t)bh * Lc * HDc;

  // Q A-fragments over the 128-deep head dim (4 chunks of K=32)
  F16x16 qf[4];
  const _Float16* qr = Qb + (size_t)(q0 + row) * HDc;
#pragma unroll
  for (int c = 0; c < 4; ++c) {
    qf[c].h[0] = *(const h8*)(qr + c * 32 + hi * 8);
    qf[c].h[1] = *(const h8*)(qr + c * 32 + 16 + hi * 8);
  }

  float mrow[8], lrow[8];
  v8f o[8];
#pragma unroll
  for (int r = 0; r < 8; ++r) { mrow[r] = -1e30f; lrow[r] = 0.f; }
#pragma unroll
  for (int tI = 0; tI < 8; ++tI) o[tI] = (v8f){};

  // stage K (128x4 segs) + V (32x16 segs): 8 async b128 per thread per buffer
#define FA_STAGE(buf, j0)                                                      \
  _Pragma("unroll")                                                            \
  for (int t = 0; t < 4; ++t) {                                                \
    int s = tid + t * 128;                                                     \
    ASYNC_B128(&Ks[buf][s >> 2][(s & 3) * 8],                                  \
               Ktb + (size_t)(s >> 2) * Lc + (j0) + (s & 3) * 8);              \
  }                                                                            \
  _Pragma("unroll")                                                            \
  for (int t = 0; t < 4; ++t) {                                                \
    int s = tid + t * 128;                                                     \
    ASYNC_B128(&Vs[buf][s >> 4][(s & 15) * 8],                                 \
               Vb + (size_t)((j0) + (s >> 4)) * HDc + (s & 15) * 8);           \
  }

  const int NIT = Lc / 32;                           // 32 key steps
  FA_STAGE(0, 0)
  for (int it = 0; it < NIT; ++it) {
    const int buf = it & 1;
    if (it + 1 < NIT) { FA_STAGE(buf ^ 1, (it + 1) * 32) WAIT_ASYNC(8); }
    else              { WAIT_ASYNC(0); }
    __syncthreads();

    // ---- S = Q . K^T from LDS (two 16x16 C-frags over 32 keys) ----
    v8f s0 = {}, s1 = {};
#pragma unroll
    for (int c = 0; c < 4; ++c) {
      const _Float16* kb = &Ks[buf][c * 32][row];
      v16h b0, b1;
#pragma unroll
      for (int s = 0; s < 16; ++s) {
        const int kk = (s < 8) ? hi * 8 + s : 16 + hi * 8 + (s - 8);
        b0[s] = kb[kk * 32];
        b1[s] = kb[kk * 32 + 16];
      }
      s0 = wmma_f16(qf[c].v, b0, s0);
      s1 = wmma_f16(qf[c].v, b1, s1);
    }

    // ---- online softmax across 16-lane column groups ----
    float mt[8], p0[8], p1[8], rs[8];
#pragma unroll
    for (int r = 0; r < 8; ++r) {
      float a = s0[r] * SCALEc, bb = s1[r] * SCALEc;
      p0[r] = a; p1[r] = bb;
      mt[r] = fmaxf(a, bb);
    }
#pragma unroll
    for (int off = 1; off < 16; off <<= 1)
#pragma unroll
      for (int r = 0; r < 8; ++r)
        mt[r] = fmaxf(mt[r], __shfl_xor(mt[r], off, 32));
#pragma unroll
    for (int r = 0; r < 8; ++r) {
      float mnew = fmaxf(mrow[r], mt[r]);
      float alpha = __expf(mrow[r] - mnew);
      p0[r] = __expf(p0[r] - mnew);
      p1[r] = __expf(p1[r] - mnew);
      rs[r] = p0[r] + p1[r];
      mrow[r] = mnew;
      lrow[r] *= alpha;
#pragma unroll
      for (int tI = 0; tI < 8; ++tI) o[tI][r] *= alpha;
    }
#pragma unroll
    for (int off = 1; off < 16; off <<= 1)
#pragma unroll
      for (int r = 0; r < 8; ++r)
        rs[r] += __shfl_xor(rs[r], off, 32);
#pragma unroll
    for (int r = 0; r < 8; ++r) lrow[r] += rs[r];

    // ---- P: C-layout -> A-layout via per-wave LDS tile ----
#pragma unroll
    for (int r = 0; r < 8; ++r) {
      Ps[wave][hi * 8 + r][col]      = (_Float16)p0[r];
      Ps[wave][hi * 8 + r][16 + col] = (_Float16)p1[r];
    }
    F16x16 pf;
    pf.h[0] = *(const h8*)&Ps[wave][row][hi * 8];
    pf.h[1] = *(const h8*)&Ps[wave][row][16 + hi * 8];

    // ---- O += P(16x32) @ V(32x128) from LDS ----
#pragma unroll
    for (int tI = 0; tI < 8; ++tI) {
      v16h bvv;
#pragma unroll
      for (int s = 0; s < 16; ++s)
        bvv[s] = Vs[buf][(s < 8) ? hi * 8 + s : 16 + hi * 8 + (s - 8)][tI * 16 + row];
      o[tI] = wmma_f16(pf.v, bvv, o[tI]);
    }
    __syncthreads();
  }
#undef FA_STAGE

  // ---- normalize, emit token-major f16 for the output projection ----
  float inv[8];
#pragma unroll
  for (int r = 0; r < 8; ++r) inv[r] = 1.0f / lrow[r];
  const int b = bh / NHc;
  const int h = bh % NHc;
#pragma unroll
  for (int tI = 0; tI < 8; ++tI) {
#pragma unroll
    for (int r = 0; r < 8; ++r) {
      int m = hi * 8 + r;
      O[((size_t)(b * Lc + q0 + m)) * DIMc + h * HDc + tI * 16 + col] =
          (_Float16)(o[tI][r] * inv[r]);
    }
  }
}

// ---------------- host-side orchestration ----------------
extern "C" void kernel_launch(void* const* d_in, const int* in_sizes, int n_in,
                              void* d_out, int out_size, void* d_ws, size_t ws_size,
                              hipStream_t stream) {
  const float* x   = (const float*)d_in[0];
  const float* Wq  = (const float*)d_in[1];
  const float* bq  = (const float*)d_in[2];
  const float* Wk  = (const float*)d_in[3];
  const float* bk  = (const float*)d_in[4];
  const float* Wv  = (const float*)d_in[5];
  const float* bv  = (const float*)d_in[6];
  const float* Wo  = (const float*)d_in[7];
  const float* bo  = (const float*)d_in[8];
  const float* gq  = (const float*)d_in[9];
  const float* gk  = (const float*)d_in[10];
  const float* fc  = (const float*)d_in[11];
  const float* fs  = (const float*)d_in[12];
  (void)d_in[13]; (void)in_sizes; (void)n_in; (void)out_size; (void)ws_size;

  // ---- workspace layout ----
  const size_t WELEM = (size_t)DIMc * DIMc;        // 4M halves per weight
  const size_t TELEM = (size_t)ROWS * DIMc;        // 8M token elements
  _Float16* WtQ = (_Float16*)d_ws;
  _Float16* WtK = WtQ + WELEM;
  _Float16* WtV = WtK + WELEM;
  _Float16* WtO = WtV + WELEM;
  _Float16* Xh  = WtO + WELEM;
  float*    Yf  = (float*)(Xh + TELEM);
  _Float16* Qh  = (_Float16*)(Yf + TELEM);
  _Float16* Kth = Qh + TELEM;
  _Float16* Vh  = Kth + TELEM;
  _Float16* Oh  = Vh + TELEM;
  float* out    = (float*)d_out;

  const int cvtBlocks  = (int)(WELEM / 256);        // 16384
  const int tokBlocks  = (int)(TELEM / 256);        // 32768
  const int gemmBlocks = (ROWS / 32) * (DIMc / 128);// 2048
  const int faBlocks   = Bc * NHc * (Lc / 64);      // 1024

  // 1) weights -> transposed f16
  wtrans_kernel<<<cvtBlocks, 256, 0, stream>>>(Wq, WtQ);
  wtrans_kernel<<<cvtBlocks, 256, 0, stream>>>(Wk, WtK);
  wtrans_kernel<<<cvtBlocks, 256, 0, stream>>>(Wv, WtV);
  wtrans_kernel<<<cvtBlocks, 256, 0, stream>>>(Wo, WtO);
  // 2) x -> f16
  convert_x_kernel<<<tokBlocks, 256, 0, stream>>>(x, Xh);

  // 3) Q path: gemm -> rmsnorm+rope -> Qh
  wmma_gemm_kernel<<<gemmBlocks, 128, 0, stream>>>(Xh, WtQ, bq, Yf);
  rmsnorm_rope_kernel<<<ROWS, 256, 0, stream>>>(Yf, gq, fc, fs, Qh, 0);
  // 4) K path: gemm -> rmsnorm+rope -> Kth (transposed)
  wmma_gemm_kernel<<<gemmBlocks, 128, 0, stream>>>(Xh, WtK, bk, Yf);
  rmsnorm_rope_kernel<<<ROWS, 256, 0, stream>>>(Yf, gk, fc, fs, Kth, 1);
  // 5) V path: gemm -> reorder -> Vh
  wmma_gemm_kernel<<<gemmBlocks, 128, 0, stream>>>(Xh, WtV, bv, Yf);
  reorder_v_kernel<<<tokBlocks, 256, 0, stream>>>(Yf, Vh);

  // 6) flash attention -> Oh (token-major f16)
  flash_attn_kernel<<<faBlocks, 128, 0, stream>>>(Qh, Kth, Vh, Oh);

  // 7) output projection -> f32 d_out
  wmma_gemm_kernel<<<gemmBlocks, 128, 0, stream>>>(Oh, WtO, bo, out);
}
